// Mlp_7_Layer_6227702579398
// MI455X (gfx1250) — compile-verified, matrix operand order checked by reading, not run on hardware
//
#include <hip/hip_runtime.h>
#include <hip/hip_bf16.h>
#include <stdint.h>

#define NFIELDS 16
#define VOCAB   10000
#define EMBED   64
#define TTASKS  10
#define BATCH   8192

typedef __attribute__((ext_vector_type(16))) __bf16 v16bf;
typedef __attribute__((ext_vector_type(8)))  float  v8f;

__device__ __forceinline__ unsigned short f2bu(float f) {
    union { float f; uint32_t u; } c; c.f = f;
    uint32_t u = c.u;
    uint32_t r = (u + 0x7FFFu + ((u >> 16) & 1u)) >> 16;  // round-to-nearest-even bf16
    return (unsigned short)r;
}
__device__ __forceinline__ float bu2f(unsigned short s) {
    union { uint32_t u; float f; } c; c.u = ((uint32_t)s) << 16;
    return c.f;
}

// ---------------------------------------------------------------------------
// 1) Embedding gather: emb[b, f*64+e] = bf16(tables[f, x[f,b], e])
// ---------------------------------------------------------------------------
__global__ void embed_gather(const int* __restrict__ x,
                             const float* __restrict__ tables,
                             unsigned short* __restrict__ emb, int B)
{
    int idx = blockIdx.x * blockDim.x + threadIdx.x;
    if (idx >= B * 1024) return;
    int b = idx >> 10;
    int c = idx & 1023;
    int f = c >> 6;
    int e = c & 63;
    int row = x[f * B + b];
    emb[idx] = f2bu(tables[((size_t)f * VOCAB + (size_t)row) * EMBED + e]);
}

// ---------------------------------------------------------------------------
// 2) Weight transpose + fp32 -> bf16:  Wt[t][o][i] = W[t][i][o]
// ---------------------------------------------------------------------------
__global__ void wtrans(const float* __restrict__ W,
                       unsigned short* __restrict__ Wt,
                       int I, int O, long total)
{
    long idx = (long)blockIdx.x * blockDim.x + threadIdx.x;
    if (idx >= total) return;
    int  t   = (int)(idx / ((long)I * O));
    long rem = idx - (long)t * I * O;
    int  o   = (int)(rem / I);
    int  i   = (int)(rem - (long)o * I);
    Wt[idx] = f2bu(W[((long)t * I + i) * O + o]);
}

// ---------------------------------------------------------------------------
// 3) Fused  Out = bf16( relu( LN( A @ Wt^T + bias ) * g + beta ) )
//    Block: 256 threads (8 waves), computes MT=32 rows x full O.
//    Each wave owns NT 16-col tiles; each B fragment is reused for 2 row-tiles.
//    No LDS / barriers in the K loop -> loads pipeline across WMMAs.
// ---------------------------------------------------------------------------
template<int NT>
__launch_bounds__(256)
__global__ void gemm_ln_relu(const unsigned short* __restrict__ A,   // [(t?)][Brows][I] bf16
                             const unsigned short* __restrict__ Wt,  // [T][O][I] bf16
                             const float* __restrict__ bias,         // [T][O]
                             const float* __restrict__ gam,          // [T][O]
                             const float* __restrict__ bet,          // [T][O]
                             unsigned short* __restrict__ Out,       // [T][Brows][O] bf16
                             int I, int Brows, int abcast)
{
    constexpr int NW = 8, MT = 32, MW = 2, KT = 32;   // MW row-tiles of 16
    constexpr int O  = NT * 16 * NW;

    const int t    = blockIdx.y;
    const int m0   = blockIdx.x * MT;
    const int tid  = threadIdx.x;
    const int wave = tid >> 5;
    const int lane = tid & 31;
    const int lh   = lane & 15;            // column / row within 16-group
    const int kh   = (lane >> 4) << 3;     // K sub-block base: 0 or 8
    const int kq   = kh >> 3;              // uint4 index: 0 or 1

    __shared__ float red_s[NW][MT];
    __shared__ float red_q[NW][MT];
    __shared__ float mu_s[MT];
    __shared__ float rs_s[MT];

    const unsigned short* Ap = A + ((size_t)(abcast ? 0 : t) * Brows + m0) * (size_t)I;
    const unsigned short* Wp = Wt + ((size_t)t * O + (size_t)wave * (NT * 16)) * (size_t)I;

    v8f acc[MW][NT];
#pragma unroll
    for (int mt = 0; mt < MW; ++mt)
#pragma unroll
        for (int n = 0; n < NT; ++n)
#pragma unroll
            for (int r = 0; r < 8; ++r) acc[mt][n][r] = 0.f;

    union Frag { v16bf v; uint4 q[2]; };

    // per-lane A row pointers (row = mt*16 + lh), contiguous 16B + 16B per k-step
    const unsigned short* ArowP[MW];
#pragma unroll
    for (int mt = 0; mt < MW; ++mt)
        ArowP[mt] = Ap + (size_t)(mt * 16 + lh) * I;

    for (int k0 = 0; k0 < I; k0 += KT) {
        Frag af[MW];
#pragma unroll
        for (int mt = 0; mt < MW; ++mt) {
            const uint4* ar = (const uint4*)(ArowP[mt] + k0);
            af[mt].q[0] = ar[kq];
            af[mt].q[1] = ar[kq + 2];
        }
#pragma unroll
        for (int n = 0; n < NT; ++n) {
            Frag bfr;
            const uint4* br = (const uint4*)(Wp + (size_t)(n * 16 + lh) * I + (size_t)k0);
            bfr.q[0] = br[kq];
            bfr.q[1] = br[kq + 2];
#pragma unroll
            for (int mt = 0; mt < MW; ++mt)
                acc[mt][n] = __builtin_amdgcn_wmma_f32_16x16x32_bf16(
                    false, af[mt].v, false, bfr.v, (short)0, acc[mt][n], false, false);
        }
    }

    // -------- epilogue: bias, LayerNorm(O), scale/shift, ReLU --------
    float s[MW][8], qq[MW][8];
#pragma unroll
    for (int mt = 0; mt < MW; ++mt)
#pragma unroll
        for (int r = 0; r < 8; ++r) { s[mt][r] = 0.f; qq[mt][r] = 0.f; }

    float gl[NT], bl[NT];
#pragma unroll
    for (int n = 0; n < NT; ++n) {
        int col = wave * (NT * 16) + n * 16 + lh;
        float bi = bias[t * O + col];
        gl[n] = gam[t * O + col];
        bl[n] = bet[t * O + col];
#pragma unroll
        for (int mt = 0; mt < MW; ++mt)
#pragma unroll
            for (int r = 0; r < 8; ++r) {
                float v = acc[mt][n][r] + bi;
                acc[mt][n][r] = v;
                s[mt][r]  += v;
                qq[mt][r] += v * v;
            }
    }
    // butterfly within the 16-lane group (C layout: lanes 0-15 rows r, 16-31 rows r+8)
#pragma unroll
    for (int m = 1; m <= 8; m <<= 1) {
#pragma unroll
        for (int mt = 0; mt < MW; ++mt)
#pragma unroll
            for (int r = 0; r < 8; ++r) {
                s[mt][r]  += __shfl_xor(s[mt][r],  m, 32);
                qq[mt][r] += __shfl_xor(qq[mt][r], m, 32);
            }
    }
    if (lh == 0) {
        int mb = (lane >> 4) << 3;
#pragma unroll
        for (int mt = 0; mt < MW; ++mt)
#pragma unroll
            for (int r = 0; r < 8; ++r) {
                red_s[wave][mt * 16 + mb + r] = s[mt][r];
                red_q[wave][mt * 16 + mb + r] = qq[mt][r];
            }
    }
    __syncthreads();
    if (tid < MT) {
        float ss = 0.f, sq = 0.f;
#pragma unroll
        for (int w = 0; w < NW; ++w) { ss += red_s[w][tid]; sq += red_q[w][tid]; }
        float mu  = ss / (float)O;
        float var = sq / (float)O - mu * mu;
        mu_s[tid] = mu;
        rs_s[tid] = rsqrtf(var + 1e-5f);
    }
    __syncthreads();

    const int mb = (lane >> 4) << 3;
    unsigned short* Op = Out + ((size_t)t * Brows + m0) * (size_t)O;
#pragma unroll
    for (int mt = 0; mt < MW; ++mt)
#pragma unroll
        for (int n = 0; n < NT; ++n) {
            int col = wave * (NT * 16) + n * 16 + lh;
#pragma unroll
            for (int r = 0; r < 8; ++r) {
                int m = mt * 16 + mb + r;
                float v = (acc[mt][n][r] - mu_s[m]) * rs_s[m] * gl[n] + bl[n];
                v = fmaxf(v, 0.f);
                Op[(size_t)m * O + col] = f2bu(v);
            }
        }
}

// ---------------------------------------------------------------------------
// 4) Final tower (128->1) + sigmoid + domain select. Reference: idx = d + 2*t.
// ---------------------------------------------------------------------------
__global__ void tower2_select(const unsigned short* __restrict__ H1, // [T][B][128] bf16
                              const float* __restrict__ hW2,        // [T][128][1]
                              const float* __restrict__ hb2,        // [T][1]
                              const int* __restrict__ x,
                              float* __restrict__ out, int B)
{
    int b = blockIdx.x * blockDim.x + threadIdx.x;
    if (b >= B) return;
    int d = x[(NFIELDS - 1) * B + b];
#pragma unroll
    for (int t = 0; t < 2; ++t) {
        int idx = d + 2 * t;
        const unsigned short* h = H1 + ((size_t)idx * B + b) * 128;
        const float* w = hW2 + (size_t)idx * 128;
        float sacc = hb2[idx];
#pragma unroll 4
        for (int i = 0; i < 128; ++i) sacc += bu2f(h[i]) * w[i];
        out[b * 2 + t] = 1.f / (1.f + expf(-sacc));
    }
}

// ---------------------------------------------------------------------------
// launch
// ---------------------------------------------------------------------------
extern "C" void kernel_launch(void* const* d_in, const int* in_sizes, int n_in,
                              void* d_out, int out_size, void* d_ws, size_t ws_size,
                              hipStream_t stream)
{
    (void)in_sizes; (void)n_in; (void)out_size; (void)ws_size;

    const int*   x      = (const int*)d_in[0];
    const float* tables = (const float*)d_in[1];
    const float* W[4]   = {(const float*)d_in[2],  (const float*)d_in[3],
                           (const float*)d_in[4],  (const float*)d_in[5]};
    const float* bs[4]  = {(const float*)d_in[6],  (const float*)d_in[7],
                           (const float*)d_in[8],  (const float*)d_in[9]};
    const float* gs[4]  = {(const float*)d_in[10], (const float*)d_in[11],
                           (const float*)d_in[12], (const float*)d_in[13]};
    const float* be[4]  = {(const float*)d_in[14], (const float*)d_in[15],
                           (const float*)d_in[16], (const float*)d_in[17]};
    const float* hW1 = (const float*)d_in[18];
    const float* hb1 = (const float*)d_in[19];
    const float* hg  = (const float*)d_in[20];
    const float* hbe = (const float*)d_in[21];
    const float* hW2 = (const float*)d_in[22];
    const float* hb2 = (const float*)d_in[23];

    char* wsb = (char*)d_ws;
    size_t off = 0;
    auto carve = [&](size_t bytes) -> char* {
        char* p = wsb + off;
        off += (bytes + 255) & ~(size_t)255;
        return p;
    };
    unsigned short* emb  = (unsigned short*)carve((size_t)BATCH * 1024 * 2);
    unsigned short* actA = (unsigned short*)carve((size_t)TTASKS * BATCH * 1024 * 2);
    unsigned short* actB = (unsigned short*)carve((size_t)TTASKS * BATCH * 512 * 2);
    unsigned short* Wt0  = (unsigned short*)carve((size_t)TTASKS * 1024 * 1024 * 2);
    unsigned short* Wt1  = (unsigned short*)carve((size_t)TTASKS * 512 * 1024 * 2);
    unsigned short* Wt2  = (unsigned short*)carve((size_t)TTASKS * 256 * 512 * 2);
    unsigned short* Wt3  = (unsigned short*)carve((size_t)TTASKS * 128 * 256 * 2);
    unsigned short* WtH1 = (unsigned short*)carve((size_t)TTASKS * 128 * 128 * 2);

    // 1) embedding gather -> bf16
    {
        int total = BATCH * 1024;
        embed_gather<<<(total + 255) / 256, 256, 0, stream>>>(x, tables, emb, BATCH);
    }
    // 2) weight transpose+convert
    auto tr = [&](const float* Wsrc, unsigned short* Wdst, int I, int O) {
        long total = (long)TTASKS * I * O;
        wtrans<<<(unsigned)((total + 255) / 256), 256, 0, stream>>>(Wsrc, Wdst, I, O, total);
    };
    tr(W[0], Wt0, 1024, 1024);
    tr(W[1], Wt1, 1024, 512);
    tr(W[2], Wt2, 512, 256);
    tr(W[3], Wt3, 256, 128);
    tr(hW1,  WtH1, 128, 128);

    // 3) fused GEMM chain (32 rows per block)
    dim3 grid(BATCH / 32, TTASKS);
    gemm_ln_relu<8><<<grid, 256, 0, stream>>>(emb,  Wt0,  bs[0], gs[0], be[0], actA, 1024, BATCH, 1);
    gemm_ln_relu<4><<<grid, 256, 0, stream>>>(actA, Wt1,  bs[1], gs[1], be[1], actB, 1024, BATCH, 0);
    gemm_ln_relu<2><<<grid, 256, 0, stream>>>(actB, Wt2,  bs[2], gs[2], be[2], actA,  512, BATCH, 0);
    gemm_ln_relu<1><<<grid, 256, 0, stream>>>(actA, Wt3,  bs[3], gs[3], be[3], actB,  256, BATCH, 0);
    gemm_ln_relu<1><<<grid, 256, 0, stream>>>(actB, WtH1, hb1,   hg,    hbe,   actA,  128, BATCH, 0);

    // 4) final tower + sigmoid + domain select
    tower2_select<<<(BATCH + 255) / 256, 256, 0, stream>>>(actA, hW2, hb2, x, (float*)d_out, BATCH);
}
